// MBPertTS_32040456028998
// MI455X (gfx1250) — matrix operand order repaired and checked by reading, not compiled, not run on hardware
//
#include <hip/hip_runtime.h>
#include <math.h>

// ---------------------------------------------------------------------------
// Problem constants (match the JAX reference)
// ---------------------------------------------------------------------------
#define NSP   8192          // n_species
#define KP    4             // perturbation count
#define TGRID 64            // P has TGRID+1 rows, idx = clip(floor(t),0,TGRID)
#define MAX_STEPS 64

// Dormand-Prince error coefficients (b5 - b4)
#define E1f (71.0f/57600.0f)
#define E3f (-71.0f/16695.0f)
#define E4f (71.0f/1920.0f)
#define E5f (-17253.0f/339200.0f)
#define E6f (22.0f/525.0f)
#define E7f (-1.0f/40.0f)

typedef __bf16          v16bf __attribute__((ext_vector_type(16)));
typedef float           v8f   __attribute__((ext_vector_type(8)));
typedef unsigned short  us8   __attribute__((ext_vector_type(8)));

__device__ __forceinline__ unsigned short f2bf(float f) {
    unsigned u = __float_as_uint(f);
    unsigned r = u + 0x7FFFu + ((u >> 16) & 1u);   // round-to-nearest-even
    return (unsigned short)(r >> 16);
}

__device__ __forceinline__ float heff_of(float t, float h, float t2) {
    return fmaxf(fminf(h, t2 - t), 1e-7f);
}

// ---------------------------------------------------------------------------
// One-time fp32 -> bf16 conversion of A into workspace (L2-resident copy)
// ---------------------------------------------------------------------------
__global__ __launch_bounds__(256) void convA_kernel(const float* __restrict__ A,
                                                    unsigned short* __restrict__ Ab) {
    size_t i = ((size_t)blockIdx.x * 256 + threadIdx.x) * 8;
    float4 f0 = *(const float4*)(A + i);
    float4 f1 = *(const float4*)(A + i + 4);
    us8 o;
    o[0] = f2bf(f0.x); o[1] = f2bf(f0.y); o[2] = f2bf(f0.z); o[3] = f2bf(f0.w);
    o[4] = f2bf(f1.x); o[5] = f2bf(f1.y); o[6] = f2bf(f1.z); o[7] = f2bf(f1.w);
    *(us8*)(Ab + i) = o;
}

// ---------------------------------------------------------------------------
// Init: y = x, stage-1 vector = x (fp32 + bf16), scalar state {t, h}
// ---------------------------------------------------------------------------
__global__ __launch_bounds__(256) void init_kernel(const float* __restrict__ x,
                                                   const float* __restrict__ t1p,
                                                   const float* __restrict__ t2p,
                                                   float* __restrict__ y,
                                                   float* __restrict__ ysf,
                                                   unsigned short* __restrict__ ysb,
                                                   float* __restrict__ sc) {
    int i = blockIdx.x * 256 + threadIdx.x;
    float v = x[i];
    y[i] = v; ysf[i] = v; ysb[i] = f2bf(v);
    if (i == 0) { sc[0] = t1p[0]; sc[1] = (t2p[0] - t1p[0]) / 16.0f; }
}

// ---------------------------------------------------------------------------
// Stage vector build: y_stage = y + h_eff * sum(a_sj * k_j)   (stages 2..7)
// Stage 7 writes y5 (== ysf), used later by control_kernel.
// ---------------------------------------------------------------------------
__global__ __launch_bounds__(256) void prep_kernel(const float* __restrict__ y,
                                                   const float* __restrict__ kbuf,
                                                   float* __restrict__ ysf,
                                                   unsigned short* __restrict__ ysb,
                                                   const float* __restrict__ sc,
                                                   const float* __restrict__ t2p,
                                                   int stage) {
    int i = blockIdx.x * 256 + threadIdx.x;
    float he = heff_of(sc[0], sc[1], *t2p);
    const float* k1 = kbuf;
    const float* k2 = kbuf + 1 * NSP;
    const float* k3 = kbuf + 2 * NSP;
    const float* k4 = kbuf + 3 * NSP;
    const float* k5 = kbuf + 4 * NSP;
    const float* k6 = kbuf + 5 * NSP;
    float acc;
    switch (stage) {
    case 2: acc = 0.2f * k1[i]; break;
    case 3: acc = 0.075f * k1[i] + 0.225f * k2[i]; break;
    case 4: acc = (44.0f/45.0f)*k1[i] - (56.0f/15.0f)*k2[i] + (32.0f/9.0f)*k3[i]; break;
    case 5: acc = (19372.0f/6561.0f)*k1[i] - (25360.0f/2187.0f)*k2[i]
                + (64448.0f/6561.0f)*k3[i] - (212.0f/729.0f)*k4[i]; break;
    case 6: acc = (9017.0f/3168.0f)*k1[i] - (355.0f/33.0f)*k2[i]
                + (46732.0f/5247.0f)*k3[i] + (49.0f/176.0f)*k4[i]
                - (5103.0f/18656.0f)*k5[i]; break;
    default: // 7 -> y5
            acc = (35.0f/384.0f)*k1[i] + (500.0f/1113.0f)*k3[i]
                + (125.0f/192.0f)*k4[i] - (2187.0f/6784.0f)*k5[i]
                + (11.0f/84.0f)*k6[i]; break;
    }
    float ys = y[i] + he * acc;
    ysf[i] = ys;
    ysb[i] = f2bf(ys);
}

// ---------------------------------------------------------------------------
// WMMA bf16 GEMV + fused GLV right-hand side:
//   k_s[row] = y_stage[row] * ( r[row] + (A @ y_stage)[row] + eps[row,:]·P[idx,:] )
// Block: 256 thr = 8 waves; 16 rows/block (one WMMA M-tile), each wave covers
// a 1024-column slice; cross-wave reduce in LDS. 512 blocks -> 4096 waves.
// A operand in contiguous K order (B uses the same): lane<16 holds K {0..15},
// lane>=16 holds K {16..31} (two contiguous 16B loads).
// B columns are all identical (= x chunk) so D columns are identical.
// ---------------------------------------------------------------------------
template <bool ABF16>
__global__ __launch_bounds__(256) void gemv_stage_kernel(
        const void* __restrict__ Aptr,
        const unsigned short* __restrict__ ysb,
        const float* __restrict__ ysf,
        const float* __restrict__ rvec,
        const float* __restrict__ eps,
        const float* __restrict__ Pm,
        const float* __restrict__ sc,
        const float* __restrict__ t2p,
        float* __restrict__ kout,
        float cstage) {
    __shared__ float part[8][16];
    const int tid  = threadIdx.x;
    const int wave = tid >> 5;
    const int lane = tid & 31;
    const int rowBase = blockIdx.x * 16;
    const int aRow    = rowBase + (lane & 15);
    const int colBase = wave * 1024;
    const int aOff = (lane < 16) ? 0 : 16;
    const int bOff = (lane < 16) ? 0 : 16;

    v8f c = {};
    if (ABF16) {
        const unsigned short* Ab =
            (const unsigned short*)Aptr + (size_t)aRow * NSP + colBase;
        #pragma unroll 4
        for (int ch = 0; ch < 1024; ch += 32) {
            union { v16bf v; us8 h[2]; } av;
            av.h[0] = *(const us8*)(Ab + ch + aOff);
            av.h[1] = *(const us8*)(Ab + ch + 8 + aOff);
            v16bf b = *(const v16bf*)(ysb + colBase + ch + bOff);
            c = __builtin_amdgcn_wmma_f32_16x16x32_bf16(
                    false, av.v, false, b, (short)0, c, false, false);
        }
    } else {
        const float* Af = (const float*)Aptr + (size_t)aRow * NSP + colBase;
        #pragma unroll 2
        for (int ch = 0; ch < 1024; ch += 32) {
            union { v16bf v; unsigned short s[16]; } av;
            #pragma unroll
            for (int j = 0; j < 8; ++j) {
                av.s[j]     = f2bf(Af[ch + aOff + j]);
                av.s[8 + j] = f2bf(Af[ch + 8 + aOff + j]);
            }
            v16bf b = *(const v16bf*)(ysb + colBase + ch + bOff);
            c = __builtin_amdgcn_wmma_f32_16x16x32_bf16(
                    false, av.v, false, b, (short)0, c, false, false);
        }
    }

    // D layout: lane<16, VGPR r -> (M=r, N=lane); lane>=16 -> (M=8+r).
    // All columns equal, so lanes 0 and 16 carry all 16 row sums.
    if ((lane & 15) == 0) {
        const int mb = (lane >> 4) * 8;
        #pragma unroll
        for (int j = 0; j < 8; ++j) part[wave][mb + j] = c[j];
    }
    __syncthreads();

    if (tid < 16) {
        float s = 0.0f;
        #pragma unroll
        for (int w = 0; w < 8; ++w) s += part[w][tid];
        const int row = rowBase + tid;
        const float t = sc[0], h = sc[1], t2 = *t2p;
        const float he = heff_of(t, h, t2);
        const float ts = t + cstage * he;          // T*ts/INTEGRATE_END == ts
        int idx = (int)floorf(ts);
        idx = idx < 0 ? 0 : (idx > TGRID ? TGRID : idx);
        const float pert = eps[row*KP+0]*Pm[idx*KP+0] + eps[row*KP+1]*Pm[idx*KP+1]
                         + eps[row*KP+2]*Pm[idx*KP+2] + eps[row*KP+3]*Pm[idx*KP+3];
        kout[row] = ysf[row] * (rvec[row] + s + pert);
    }
}

// ---------------------------------------------------------------------------
// Step control: RMS error, accept/step-size update, masked state update.
// Also refreshes (ysf, ysb) = new y so the next step's stage 1 needs no prep.
// ---------------------------------------------------------------------------
__global__ __launch_bounds__(256) void control_kernel(float* __restrict__ y,
                                                      float* __restrict__ sc,
                                                      const float* __restrict__ kbuf,
                                                      float* __restrict__ ysf,   // holds y5
                                                      unsigned short* __restrict__ ysb,
                                                      const float* __restrict__ t2p) {
    __shared__ float red[256];
    __shared__ float s_upd;
    const int tid = threadIdx.x;
    const float t = sc[0], h = sc[1], t2 = *t2p;
    const bool done = (t >= t2 - 1e-6f);
    const float he = heff_of(t, h, t2);
    const float* k1 = kbuf;
    const float* k3 = kbuf + 2 * NSP;
    const float* k4 = kbuf + 3 * NSP;
    const float* k5 = kbuf + 4 * NSP;
    const float* k6 = kbuf + 5 * NSP;
    const float* k7 = kbuf + 6 * NSP;

    float acc = 0.0f;
    for (int i = tid; i < NSP; i += 256) {
        float e = he * (E1f*k1[i] + E3f*k3[i] + E4f*k4[i] + E5f*k5[i]
                        + E6f*k6[i] + E7f*k7[i]);
        float s = 1e-6f + 1e-3f * fmaxf(fabsf(y[i]), fabsf(ysf[i]));
        float q = e / s;
        acc += q * q;
    }
    red[tid] = acc;
    __syncthreads();
    for (int off = 128; off > 0; off >>= 1) {
        if (tid < off) red[tid] += red[tid + off];
        __syncthreads();
    }
    if (tid == 0) {
        float err_norm = sqrtf(red[0] / (float)NSP);
        bool accept = (err_norm <= 1.0f);
        float safe = fmaxf(err_norm, 1e-10f);
        float factor = fminf(fmaxf(0.9f * powf(safe, -0.2f), 0.2f), 10.0f);
        if (!done) {
            sc[0] = accept ? (t + he) : t;
            sc[1] = he * factor;
        }
        s_upd = (!done && accept) ? 1.0f : 0.0f;
    }
    __syncthreads();
    const bool upd = (s_upd != 0.0f);
    for (int i = tid; i < NSP; i += 256) {
        float yn = upd ? ysf[i] : y[i];
        y[i] = yn;
        ysf[i] = yn;           // next step's stage-1 vector
        ysb[i] = f2bf(yn);
    }
}

// ---------------------------------------------------------------------------
// Host driver (graph-capture safe: launches only, all on `stream`)
// ---------------------------------------------------------------------------
extern "C" void kernel_launch(void* const* d_in, const int* in_sizes, int n_in,
                              void* d_out, int out_size, void* d_ws, size_t ws_size,
                              hipStream_t stream) {
    (void)in_sizes; (void)n_in; (void)out_size;
    const float* x   = (const float*)d_in[0];
    const float* t1p = (const float*)d_in[1];
    const float* t2p = (const float*)d_in[2];
    const float* rv  = (const float*)d_in[3];
    const float* A   = (const float*)d_in[4];
    const float* eps = (const float*)d_in[5];
    const float* P   = (const float*)d_in[6];
    float* y = (float*)d_out;                      // evolving state == output

    char* ws = (char*)d_ws;
    float* sc           = (float*)ws;                              // {t, h}
    float* kbuf         = (float*)(ws + 256);                      // 7 * N f32
    float* ysf          = (float*)(ws + 256 + 7 * NSP * 4);        // N f32
    unsigned short* ysb = (unsigned short*)(ws + 256 + 8 * NSP * 4); // N bf16
    const size_t A_OFF  = 512 * 1024;
    unsigned short* Ab  = (unsigned short*)(ws + A_OFF);           // N*N bf16
    const bool cacheA   = ws_size >= A_OFF + (size_t)NSP * NSP * 2;

    init_kernel<<<NSP / 256, 256, 0, stream>>>(x, t1p, t2p, y, ysf, ysb, sc);
    if (cacheA)
        convA_kernel<<<(int)(((size_t)NSP * NSP) / (256 * 8)), 256, 0, stream>>>(A, Ab);

    static const float cst[8] = {0.f, 0.f, 0.2f, 0.3f, 0.8f, 8.f/9.f, 1.f, 1.f};
    for (int step = 0; step < MAX_STEPS; ++step) {
        for (int s = 1; s <= 7; ++s) {
            if (s > 1)
                prep_kernel<<<NSP / 256, 256, 0, stream>>>(y, kbuf, ysf, ysb, sc, t2p, s);
            float* kout = kbuf + (size_t)(s - 1) * NSP;
            if (cacheA)
                gemv_stage_kernel<true><<<NSP / 16, 256, 0, stream>>>(
                    (const void*)Ab, ysb, ysf, rv, eps, P, sc, t2p, kout, cst[s]);
            else
                gemv_stage_kernel<false><<<NSP / 16, 256, 0, stream>>>(
                    (const void*)A, ysb, ysf, rv, eps, P, sc, t2p, kout, cst[s]);
        }
        control_kernel<<<1, 256, 0, stream>>>(y, sc, kbuf, ysf, ysb, t2p);
    }
}
